// CapsuleNet_12008728559930
// MI455X (gfx1250) — matrix operand order repaired and checked by reading, not compiled, not run on hardware
//
#include <hip/hip_runtime.h>

// ---------------- CDNA5 WMMA types ----------------
typedef __attribute__((ext_vector_type(16))) __bf16 v16bf;
typedef __attribute__((ext_vector_type(8)))  float  v8f;

union Frag {            // 16 bf16 per lane = 32 bytes
  v16bf v;
  uint4 q[2];
};
union Pack8h { unsigned short s[8]; uint4 q; };
union Pack8f { float f[8]; float4 q[2]; };

__device__ __forceinline__ unsigned short f2bf(float f) {
  unsigned int u = __float_as_uint(f);
  return (unsigned short)((u + 0x7FFFu + ((u >> 16) & 1u)) >> 16);
}
__device__ __forceinline__ float bf2f(unsigned short h) {
  return __uint_as_float(((unsigned int)h) << 16);
}

// Problem constants
#define BN 64
#define LL 512
#define EE 512
#define CC 32
#define UU 8
#define KK 9
#define SS 256
#define LP 520   // L + 8 (halo of 4 each side)

// ---------------- prep: pack weights to bf16 GEMM layouts, zero halos & logits ----
// w1 [32][4608]  k = t*512 + e   <- conv1_w [C][E][9]
// w2 [256][288]  k = t*32 + c    <- prim_w  [UC][C][9]
// Wt [u][k][d][c]                <- W [u][k][c][d]
__global__ void k_prep(const float* __restrict__ conv1_w,
                       const float* __restrict__ prim_w,
                       const float* __restrict__ Wm,
                       unsigned short* __restrict__ w1,
                       unsigned short* __restrict__ w2,
                       unsigned short* __restrict__ Wt,
                       unsigned short* __restrict__ e_pad,
                       unsigned short* __restrict__ h_pad,
                       float* __restrict__ b_l) {
  int tid = blockIdx.x * blockDim.x + threadIdx.x;
  if (tid < 32 * 4608) {
    int c = tid / 4608, k = tid % 4608;
    int t = k >> 9, e = k & 511;
    w1[tid] = f2bf(conv1_w[(c * 512 + e) * 9 + t]);
    return;
  }
  tid -= 32 * 4608;
  if (tid < 256 * 288) {
    int uc = tid / 288, k = tid % 288;
    int t = k >> 5, c = k & 31;
    w2[tid] = f2bf(prim_w[(uc * 32 + c) * 9 + t]);
    return;
  }
  tid -= 256 * 288;
  if (tid < UU * KK * 32 * 32) {
    int c = tid & 31, d = (tid >> 5) & 31, uk = tid >> 10;
    Wt[tid] = f2bf(Wm[(uk * 32 + c) * 32 + d]);
    return;
  }
  tid -= UU * KK * 32 * 32;
  if (tid < BN * 8 * EE) {            // e_pad halo rows: 0..3 and 516..519
    int b = tid / (8 * EE), r = (tid / EE) % 8, e = tid & (EE - 1);
    int row = (r < 4) ? r : (516 + r - 4);
    e_pad[((size_t)b * LP + row) * EE + e] = 0;
    return;
  }
  tid -= BN * 8 * EE;
  if (tid < BN * 8 * CC) {            // h_pad halo rows
    int b = tid / (8 * CC), r = (tid / CC) % 8, c = tid & (CC - 1);
    int row = (r < 4) ? r : (516 + r - 4);
    h_pad[((size_t)b * LP + row) * CC + c] = 0;
    return;
  }
  tid -= BN * 8 * CC;
  if (tid < BN * UU * KK) b_l[tid] = 0.0f;
}

// ---------------- embedding gather -> bf16 halo-padded [B][L+8][E] ----------------
__global__ void k_embed(const int* __restrict__ x, const float* __restrict__ mask,
                        const float* __restrict__ emb, unsigned short* __restrict__ e_pad) {
  int tid = blockIdx.x * blockDim.x + threadIdx.x;   // B*L*64 threads, 8 elems each
  int e0 = (tid & 63) * 8;
  int bl = tid >> 6;
  int b = bl >> 9, l = bl & 511;
  int idx = x[bl];
  float m = mask[bl];
  const float* src = emb + (size_t)idx * EE + e0;
  float4 a0 = *(const float4*)(src);
  float4 a1 = *(const float4*)(src + 4);
  Pack8h o;
  o.s[0] = f2bf(a0.x * m); o.s[1] = f2bf(a0.y * m);
  o.s[2] = f2bf(a0.z * m); o.s[3] = f2bf(a0.w * m);
  o.s[4] = f2bf(a1.x * m); o.s[5] = f2bf(a1.y * m);
  o.s[6] = f2bf(a1.z * m); o.s[7] = f2bf(a1.w * m);
  *(uint4*)(e_pad + ((size_t)b * LP + l + 4) * EE + e0) = o.q;
}

// ---------------- conv1 as WMMA GEMM: M=32(c) N=512(l) K=4608, +bias +ReLU --------
// Each wave: both M-tiles (c 0..31) for one 16-wide l tile; B frag shared.
// im2col address is LINEAR in k: patch(k) = e_pad[b][l*512 + k].
__global__ void k_conv1(const unsigned short* __restrict__ w1,
                        const unsigned short* __restrict__ e_pad,
                        const float* __restrict__ conv1_b,
                        unsigned short* __restrict__ h_pad) {
  int wave = (blockIdx.x * blockDim.x + threadIdx.x) >> 5;   // 2048 waves
  int lane = threadIdx.x & 31;
  int nt = wave & 31, b = wave >> 5;
  int half = lane >> 4, n = lane & 15;
  const unsigned short* rowA0 = w1 + (size_t)n * 4608 + half * 8;
  const unsigned short* rowA1 = w1 + (size_t)(16 + n) * 4608 + half * 8;
  int l = nt * 16 + n;
  const unsigned short* colB = e_pad + ((size_t)b * LP + l) * EE + half * 8;
  v8f acc0 = {0.f, 0.f, 0.f, 0.f, 0.f, 0.f, 0.f, 0.f};
  v8f acc1 = {0.f, 0.f, 0.f, 0.f, 0.f, 0.f, 0.f, 0.f};
#pragma unroll 4
  for (int ks = 0; ks < 144; ++ks) {
    int k0 = ks * 32;
    Frag a0, a1, bb;
    a0.q[0] = *(const uint4*)(rowA0 + k0);
    a0.q[1] = *(const uint4*)(rowA0 + k0 + 16);
    a1.q[0] = *(const uint4*)(rowA1 + k0);
    a1.q[1] = *(const uint4*)(rowA1 + k0 + 16);
    bb.q[0] = *(const uint4*)(colB + k0);
    bb.q[1] = *(const uint4*)(colB + k0 + 16);
    acc0 = __builtin_amdgcn_wmma_f32_16x16x32_bf16(false, a0.v, false, bb.v,
                                                   (short)0, acc0, false, false);
    acc1 = __builtin_amdgcn_wmma_f32_16x16x32_bf16(false, a1.v, false, bb.v,
                                                   (short)0, acc1, false, false);
  }
  unsigned short* dstrow = h_pad + ((size_t)b * LP + l + 4) * CC;
  {
    int c_base = half * 8;
    Pack8h o;
    for (int r = 0; r < 8; ++r) {
      float vv = acc0[r] + conv1_b[c_base + r];
      o.s[r] = f2bf(vv > 0.f ? vv : 0.f);
    }
    *(uint4*)(dstrow + c_base) = o.q;
  }
  {
    int c_base = 16 + half * 8;
    Pack8h o;
    for (int r = 0; r < 8; ++r) {
      float vv = acc1[r] + conv1_b[c_base + r];
      o.s[r] = f2bf(vv > 0.f ? vv : 0.f);
    }
    *(uint4*)(dstrow + c_base) = o.q;
  }
}

// ---------------- primary caps conv: M=256(uc) N=256(s) K=288, stride 2, +bias ----
// Each wave: 2 M-tiles (one u = 32 channels) x one 16-wide s tile; B frag shared.
// patch(k) = h_pad[b][s*64 + k]  (linear in k).
__global__ void k_prim(const unsigned short* __restrict__ w2,
                       const unsigned short* __restrict__ h_pad,
                       const float* __restrict__ prim_b,
                       float* __restrict__ p_raw) {
  int wave = (blockIdx.x * blockDim.x + threadIdx.x) >> 5;   // 8192 waves
  int lane = threadIdx.x & 31;
  int nt = wave & 15, u = (wave >> 4) & 7, b = wave >> 7;
  int half = lane >> 4, n = lane & 15;
  const unsigned short* rowA0 = w2 + (size_t)(u * 32 + n) * 288 + half * 8;
  const unsigned short* rowA1 = w2 + (size_t)(u * 32 + 16 + n) * 288 + half * 8;
  int s = nt * 16 + n;
  const unsigned short* colB = h_pad + ((size_t)b * LP + 2 * s) * CC + half * 8;
  v8f acc0 = {0.f, 0.f, 0.f, 0.f, 0.f, 0.f, 0.f, 0.f};
  v8f acc1 = {0.f, 0.f, 0.f, 0.f, 0.f, 0.f, 0.f, 0.f};
#pragma unroll
  for (int t = 0; t < 9; ++t) {
    int k0 = t * 32;
    Frag a0, a1, bb;
    a0.q[0] = *(const uint4*)(rowA0 + k0);
    a0.q[1] = *(const uint4*)(rowA0 + k0 + 16);
    a1.q[0] = *(const uint4*)(rowA1 + k0);
    a1.q[1] = *(const uint4*)(rowA1 + k0 + 16);
    bb.q[0] = *(const uint4*)(colB + k0);
    bb.q[1] = *(const uint4*)(colB + k0 + 16);
    acc0 = __builtin_amdgcn_wmma_f32_16x16x32_bf16(false, a0.v, false, bb.v,
                                                   (short)0, acc0, false, false);
    acc1 = __builtin_amdgcn_wmma_f32_16x16x32_bf16(false, a1.v, false, bb.v,
                                                   (short)0, acc1, false, false);
  }
  float* dst = p_raw + (((size_t)(b * UU + u) * SS) + s) * CC;
  {
    int c0 = half * 8;
    Pack8f o;
    for (int r = 0; r < 8; ++r) o.f[r] = acc0[r] + prim_b[u * 32 + c0 + r];
    *(float4*)(dst + c0) = o.q[0];
    *(float4*)(dst + c0 + 4) = o.q[1];
  }
  {
    int c0 = 16 + half * 8;
    Pack8f o;
    for (int r = 0; r < 8; ++r) o.f[r] = acc1[r] + prim_b[u * 32 + c0 + r];
    *(float4*)(dst + c0) = o.q[0];
    *(float4*)(dst + c0 + 4) = o.q[1];
  }
}

// ---------------- squash over C (lane = c), output bf16 in A-friendly [.,s,c] -----
__global__ void k_squash(const float* __restrict__ p_raw, unsigned short* __restrict__ p_bf) {
  int tid = blockIdx.x * blockDim.x + threadIdx.x;
  float xv = p_raw[tid];
  float sq = xv * xv;
  for (int off = 16; off > 0; off >>= 1) sq += __shfl_xor(sq, off, 32);
  float scale = (sq / (1.f + sq)) / sqrtf(sq + 1e-8f);
  p_bf[tid] = f2bf(xv * scale);
}

// ---------------- u_hat: per (b,u,s-tile): A=p[16s x 32c], B=Wt[u][k] -> 18 WMMA --
// out: u_hat bf16 [B][U][K][S][C]  (bf16 halves routing traffic; read 3x)
__global__ void k_uhat(const unsigned short* __restrict__ p_bf,
                       const unsigned short* __restrict__ Wt,
                       unsigned short* __restrict__ u_hat) {
  int wave = (blockIdx.x * blockDim.x + threadIdx.x) >> 5;   // 8192 waves
  int lane = threadIdx.x & 31;
  int st = wave & 15, u = (wave >> 4) & 7, b = wave >> 7;
  int half = lane >> 4, n = lane & 15;
  const unsigned short* rowA = p_bf + (((size_t)(b * UU + u) * SS) + st * 16 + n) * CC;
  Frag a;
  a.q[0] = *(const uint4*)(rowA + half * 8);
  a.q[1] = *(const uint4*)(rowA + 16 + half * 8);
  int s0 = st * 16 + half * 8;
  for (int k = 0; k < KK; ++k) {
    size_t base = (((size_t)(b * UU + u) * KK + k) * SS) * CC;
    const unsigned short* wk = Wt + ((size_t)(u * KK + k) * 32) * 32;
#pragma unroll
    for (int dt = 0; dt < 2; ++dt) {
      Frag bb;
      const unsigned short* colB = wk + (dt * 16 + n) * 32 + half * 8;
      bb.q[0] = *(const uint4*)(colB);
      bb.q[1] = *(const uint4*)(colB + 16);
      v8f acc = {0.f, 0.f, 0.f, 0.f, 0.f, 0.f, 0.f, 0.f};
      acc = __builtin_amdgcn_wmma_f32_16x16x32_bf16(false, a.v, false, bb.v,
                                                    (short)0, acc, false, false);
      int d = dt * 16 + n;
      for (int r = 0; r < 8; ++r)
        u_hat[base + (size_t)(s0 + r) * CC + d] = f2bf(acc[r]);
    }
  }
}

// ---------------- routing: softmax over K per (b,u) (b is S-independent) ----------
__global__ void k_soft(const float* __restrict__ b_l, float* __restrict__ cc) {
  int tid = blockIdx.x * blockDim.x + threadIdx.x;  // B*U = 512
  if (tid >= BN * UU) return;
  const float* src = b_l + tid * KK;
  float ex[KK], mx = src[0];
  for (int k = 1; k < KK; ++k) mx = fmaxf(mx, src[k]);
  float sum = 0.f;
  for (int k = 0; k < KK; ++k) { ex[k] = __expf(src[k] - mx); sum += ex[k]; }
  float inv = 1.f / sum;
  for (int k = 0; k < KK; ++k) cc[tid * KK + k] = ex[k] * inv;
}

// ------- routing: s = sum_u c*u_hat; v = squash(s); fused agreement (atomic) ------
// lane = c.  u_hat values are already in registers -> agree = 8 lane-reductions.
__global__ void k_route_v(const unsigned short* __restrict__ u_hat,
                          const float* __restrict__ cc,
                          float* __restrict__ v,
                          float* __restrict__ b_l,
                          int do_agree) {
  int wid = (blockIdx.x * blockDim.x + threadIdx.x) >> 5;    // B*K*S = 147456
  int lane = threadIdx.x & 31;
  int s = wid & 255, k = (wid >> 8) % KK, b = wid / (KK * SS);
  float uh[UU];
  float acc = 0.f;
#pragma unroll
  for (int u = 0; u < UU; ++u) {
    uh[u] = bf2f(u_hat[(((size_t)(b * UU + u) * KK + k) * SS + s) * CC + lane]);
    acc += cc[(b * UU + u) * KK + k] * uh[u];
  }
  float sq = acc * acc;
  for (int off = 16; off > 0; off >>= 1) sq += __shfl_xor(sq, off, 32);
  float scale = (sq / (1.f + sq)) / sqrtf(sq + 1e-8f);
  float vval = acc * scale;
  v[(((size_t)(b * KK + k) * SS) + s) * CC + lane] = vval;
  if (do_agree) {
#pragma unroll
    for (int u = 0; u < UU; ++u) {
      float ag = uh[u] * vval;
      for (int off = 16; off > 0; off >>= 1) ag += __shfl_xor(ag, off, 32);
      if (lane == 0) atomicAdd(&b_l[(b * UU + u) * KK + k], ag);
    }
  }
}

// ---------------- out[b,k,c] = mean over s of v ----------------------------------
__global__ void k_mean(const float* __restrict__ v, float* __restrict__ out) {
  int wid = (blockIdx.x * blockDim.x + threadIdx.x) >> 5;    // B*K = 576
  int lane = threadIdx.x & 31;
  int k = wid % KK, b = wid / KK;
  const float* src = v + ((size_t)(b * KK + k) * SS) * CC + lane;
  float s = 0.f;
  for (int ss = 0; ss < SS; ++ss) s += src[(size_t)ss * CC];
  out[(b * KK + k) * CC + lane] = s * (1.0f / SS);
}

// ---------------- host ------------------------------------------------------------
extern "C" void kernel_launch(void* const* d_in, const int* in_sizes, int n_in,
                              void* d_out, int out_size, void* d_ws, size_t ws_size,
                              hipStream_t stream) {
  (void)in_sizes; (void)n_in; (void)out_size; (void)ws_size;
  const int*   x       = (const int*)  d_in[0];
  const float* mask    = (const float*)d_in[1];
  const float* emb     = (const float*)d_in[2];
  const float* conv1_w = (const float*)d_in[3];
  const float* conv1_b = (const float*)d_in[4];
  const float* prim_w  = (const float*)d_in[5];
  const float* prim_b  = (const float*)d_in[6];
  const float* Wm      = (const float*)d_in[7];
  float* out = (float*)d_out;

  char* ws = (char*)d_ws;
  constexpr size_t off_e    = 0;                                      // bf16 [B][LP][E]
  constexpr size_t off_w1   = off_e    + (size_t)BN * LP * EE * 2;
  constexpr size_t off_h    = off_w1   + (size_t)32 * 4608 * 2;
  constexpr size_t off_w2   = off_h    + (size_t)BN * LP * CC * 2;
  constexpr size_t off_Wt   = off_w2   + (size_t)256 * 288 * 2;
  constexpr size_t off_praw = off_Wt   + (size_t)UU * KK * 32 * 32 * 2;
  constexpr size_t off_pbf  = off_praw + (size_t)BN * UU * SS * CC * 4;
  constexpr size_t off_uh   = off_pbf  + (size_t)BN * UU * SS * CC * 2;
  constexpr size_t off_bl   = off_uh   + (size_t)BN * UU * KK * SS * CC * 2;  // bf16
  constexpr size_t off_cc   = off_bl   + (size_t)BN * UU * KK * 4;
  constexpr size_t off_v    = off_cc   + (size_t)BN * UU * KK * 4;

  unsigned short* e_pad = (unsigned short*)(ws + off_e);
  unsigned short* w1    = (unsigned short*)(ws + off_w1);
  unsigned short* h_pad = (unsigned short*)(ws + off_h);
  unsigned short* w2    = (unsigned short*)(ws + off_w2);
  unsigned short* Wt    = (unsigned short*)(ws + off_Wt);
  float*          p_raw = (float*)(ws + off_praw);
  unsigned short* p_bf  = (unsigned short*)(ws + off_pbf);
  unsigned short* u_hat = (unsigned short*)(ws + off_uh);
  float*          b_l   = (float*)(ws + off_bl);
  float*          ccw   = (float*)(ws + off_cc);
  float*          v     = (float*)(ws + off_v);

  k_prep  <<<2258, 256, 0, stream>>>(conv1_w, prim_w, Wm, w1, w2, Wt, e_pad, h_pad, b_l);
  k_embed <<<8192, 256, 0, stream>>>(x, mask, emb, e_pad);
  k_conv1 <<<256,  256, 0, stream>>>(w1, e_pad, conv1_b, h_pad);
  k_prim  <<<1024, 256, 0, stream>>>(w2, h_pad, prim_b, p_raw);
  k_squash<<<16384,256, 0, stream>>>(p_raw, p_bf);
  k_uhat  <<<1024, 256, 0, stream>>>(p_bf, Wt, u_hat);
  for (int r = 0; r < 3; ++r) {
    k_soft   <<<2,     256, 0, stream>>>(b_l, ccw);
    k_route_v<<<18432, 256, 0, stream>>>(u_hat, ccw, v, b_l, (r < 2) ? 1 : 0);
  }
  k_mean<<<72, 256, 0, stream>>>(v, out);
}
